// SelfAttention2d_60765197303891
// MI455X (gfx1250) — matrix operand order repaired and checked by reading, not compiled, not run on hardware
//
#include <hip/hip_runtime.h>
#include <hip/hip_bf16.h>

typedef __attribute__((ext_vector_type(16))) _Float16 v16h;
typedef __attribute__((ext_vector_type(8)))  _Float16 v8h;
typedef __attribute__((ext_vector_type(8)))  float    v8f;

#define B_   32
#define C_   256
#define HW_  1024
#define G_   8

__device__ inline v8f zero8() {
    v8f z = {0.f,0.f,0.f,0.f,0.f,0.f,0.f,0.f};
    return z;
}

// ---- CDNA5 async global->LDS copy (ASYNCcnt-tracked), per-lane 16B ----
__device__ inline unsigned lds_off32(const void* p) {
    // generic pointer to __shared__: addr[31:0] is the LDS offset (ISA 10.2)
    return (unsigned)(unsigned long long)p;
}
__device__ inline void async_load16(unsigned lds_off, const void* gaddr) {
    asm volatile("global_load_async_to_lds_b128 %0, %1, off"
                 :: "v"(lds_off), "v"(gaddr) : "memory");
}
__device__ inline void wait_async0() {
    asm volatile("s_wait_asynccnt 0" ::: "memory");
}

// A-fragment (16x32 f16, M x K), source row-major [rows][ld], K contiguous.
// lane<16: row0+lr, K k0..k0+7 (v0-3), k0+16..k0+23 (v4-7); lane>=16: K+8 / K+24
__device__ inline v16h frag_a(const _Float16* p, int ld, int row0, int k0, int lane) {
    int half = lane >> 4, lr = lane & 15;
    const _Float16* base = p + (size_t)(row0 + lr) * ld + k0 + half * 8;
    v8h lo = *(const v8h*)base;
    v8h hi = *(const v8h*)(base + 16);
    v16h r;
#pragma unroll
    for (int i = 0; i < 8; ++i) { r[i] = lo[i]; r[i + 8] = hi[i]; }
    return r;
}

// B-fragment (32x16 f16, K x N), source stored [N][ld] with K contiguous.
// lane<16: col0+lr, K k0..k0+7 (v0-3), k0+16..k0+23 (v4-7); lane>=16: K+8 / K+24
__device__ inline v16h frag_b(const _Float16* p, int ld, int col0, int k0, int lane) {
    int half = lane >> 4, lr = lane & 15;
    const _Float16* base = p + (size_t)(col0 + lr) * ld + k0 + half * 8;
    v8h lo = *(const v8h*)base;
    v8h hi = *(const v8h*)(base + 16);
    v16h r;
#pragma unroll
    for (int i = 0; i < 8; ++i) { r[i] = lo[i]; r[i + 8] = hi[i]; }
    return r;
}

__device__ inline v8f wmma_f16(v16h a, v16h b, v8f c) {
    return __builtin_amdgcn_wmma_f32_16x16x32_f16(false, a, false, b, (short)0, c, false, false);
}

// ---------------- K0: weights fp32 -> fp16 ----------------
__global__ void cvt_weights_kernel(const float* __restrict__ wqkv, const float* __restrict__ wproj,
                                   _Float16* __restrict__ wq16, _Float16* __restrict__ wp16) {
    int i = blockIdx.x * 256 + threadIdx.x;
    if (i < 3 * C_ * C_) wq16[i] = (_Float16)wqkv[i];
    if (i < C_ * C_)     wp16[i] = (_Float16)wproj[i];
}

// ---------------- K1: GroupNorm stats ----------------
__global__ void gn_stats_kernel(const float* __restrict__ x, float* __restrict__ mean,
                                float* __restrict__ rstd) {
    int bg = blockIdx.x;                       // b*8+g ; group = 32 channels * 1024 px
    const float* p = x + (size_t)bg * 32 * HW_;
    float s = 0.f, s2 = 0.f;
    for (int i = threadIdx.x; i < 32 * HW_; i += 256) {
        float v = p[i]; s += v; s2 += v * v;
    }
    __shared__ float sm[256], sq[256];
    sm[threadIdx.x] = s; sq[threadIdx.x] = s2;
    __syncthreads();
    for (int off = 128; off > 0; off >>= 1) {
        if (threadIdx.x < off) { sm[threadIdx.x] += sm[threadIdx.x + off];
                                 sq[threadIdx.x] += sq[threadIdx.x + off]; }
        __syncthreads();
    }
    if (threadIdx.x == 0) {
        float m = sm[0] * (1.f / 32768.f);
        float var = sq[0] * (1.f / 32768.f) - m * m;
        mean[bg] = m;
        rstd[bg] = rsqrtf(var + 1e-5f);
    }
}

// ---------------- K2: GN apply + transpose -> xnT[B][HW][C] f16 ----------------
__global__ void gn_apply_kernel(const float* __restrict__ x, const float* __restrict__ gn_w,
                                const float* __restrict__ gn_b, const float* __restrict__ mean,
                                const float* __restrict__ rstd, _Float16* __restrict__ xnT) {
    __shared__ float tile[64][65];
    int hw0 = blockIdx.x * 64, c0 = blockIdx.y * 64, b = blockIdx.z;
    int t = threadIdx.x;
    int hwl = t & 63, cb = t >> 6;
    const float* xb = x + (size_t)b * C_ * HW_;
    for (int ci = cb; ci < 64; ci += 4) {
        int c = c0 + ci;
        int g = c >> 5;
        float m = mean[b * G_ + g], r = rstd[b * G_ + g];
        float sc = gn_w[c] * r;
        float sh = gn_b[c] - m * sc;
        tile[ci][hwl] = xb[(size_t)c * HW_ + hw0 + hwl] * sc + sh;
    }
    __syncthreads();
    int cl = t & 63, hb = t >> 6;
    _Float16* dst = xnT + (size_t)b * HW_ * C_;
    for (int hi = hb; hi < 64; hi += 4)
        dst[(size_t)(hw0 + hi) * C_ + c0 + cl] = (_Float16)tile[cl][hi];
}

// ---------------- K3: QKV GEMM (WMMA, async-staged B tile) ----------------
__global__ __launch_bounds__(128) void qkv_gemm_kernel(
        const _Float16* __restrict__ w16, const _Float16* __restrict__ xnT,
        const float* __restrict__ bias, _Float16* __restrict__ qT,
        _Float16* __restrict__ kT, _Float16* __restrict__ vM) {
    int tid = threadIdx.x;
    int lane = tid & 31, wid = tid >> 5;
    int half = lane >> 4, lr = lane & 15;
    int n0 = blockIdx.x * 64;
    int m0 = blockIdx.y * 64 + wid * 16;       // output-channel tile (0..767)
    int b  = blockIdx.z;
    const _Float16* xb = xnT + (size_t)b * HW_ * C_;

    __shared__ __align__(16) _Float16 Bt[64][C_];          // 32 KB, shared by 4 waves
    {
        unsigned dst = lds_off32(&Bt[0][0]);
        const char* src = (const char*)(xb + (size_t)n0 * C_);   // contiguous 32 KB
#pragma unroll
        for (int it = 0; it < 16; ++it) {
            int off = (tid + it * 128) * 16;
            async_load16(dst + off, src + off);
        }
    }
    wait_async0();
    __syncthreads();

    v8f acc[4];
#pragma unroll
    for (int f = 0; f < 4; ++f) acc[f] = zero8();

#pragma unroll
    for (int k0 = 0; k0 < C_; k0 += 32) {
        v16h a = frag_a(w16, C_, m0, k0, lane);
        v16h bb[4];
#pragma unroll
        for (int f = 0; f < 4; ++f) bb[f] = frag_b(&Bt[0][0], C_, f * 16, k0, lane);
#pragma unroll
        for (int f = 0; f < 4; ++f) acc[f] = wmma_f16(a, bb[f], acc[f]);
    }

    float bs[8];
#pragma unroll
    for (int r = 0; r < 8; ++r) bs[r] = bias[m0 + 8 * half + r];

    int mt = m0 >> 8;                           // 0=q 1=k 2=v
    if (mt < 2) {
        _Float16* dst = (mt == 0 ? qT : kT) + (size_t)b * HW_ * C_;
        int o0 = (m0 & 255) + 8 * half;
#pragma unroll
        for (int f = 0; f < 4; ++f) {
            int n = n0 + f * 16 + lr;
#pragma unroll
            for (int r = 0; r < 8; ++r)
                dst[(size_t)n * C_ + o0 + r] = (_Float16)(acc[f][r] + bs[r]);
        }
    } else {
        _Float16* dst = vM + (size_t)b * C_ * HW_;
        int c0 = (m0 & 255) + 8 * half;
#pragma unroll
        for (int f = 0; f < 4; ++f) {
            int n = n0 + f * 16 + lr;
#pragma unroll
            for (int r = 0; r < 8; ++r)
                dst[(size_t)(c0 + r) * HW_ + n] = (_Float16)(acc[f][r] + bs[r]);
        }
    }
}

// ---------------- K4: fused flash attention, double-buffered async K/V ----------------
__global__ __launch_bounds__(128) void attention_kernel(
        const _Float16* __restrict__ qT, const _Float16* __restrict__ kT,
        const _Float16* __restrict__ vM, _Float16* __restrict__ outT) {
    int tid = threadIdx.x;
    int lane = tid & 31, wid = tid >> 5;
    int half = lane >> 4, lr = lane & 15;
    int b   = blockIdx.y;
    int ib0 = blockIdx.x * 64;                  // block query base; wave handles ib0+wid*16

    const _Float16* q  = qT + (size_t)b * HW_ * C_;
    const _Float16* k  = kT + (size_t)b * HW_ * C_;
    const _Float16* vv = vM + (size_t)b * C_ * HW_;

    __shared__ __align__(16) _Float16 Qt[64][C_];          // 32 KB (staged once)
    __shared__ __align__(16) _Float16 Kt[2][32][C_];       // 2x16 KB double buffer
    __shared__ __align__(16) _Float16 Vt[2][C_][32];       // 2x16 KB double buffer
    __shared__ __align__(16) _Float16 Plds[4][16][32];     // 4 KB

    // stage Q tile (contiguous 32 KB in [HW][C])
    {
        unsigned dst = lds_off32(&Qt[0][0]);
        const char* src = (const char*)(q + (size_t)ib0 * C_);
#pragma unroll
        for (int it = 0; it < 16; ++it) {
            int off = (tid + it * 128) * 16;
            async_load16(dst + off, src + off);
        }
    }
    // stage K/V tile for j0 into buffer `buf`
    auto stage_kv = [&](int buf, int j0) {
        unsigned dstK = lds_off32(&Kt[buf][0][0]);
        const char* srcK = (const char*)(k + (size_t)j0 * C_);   // contiguous 16 KB
#pragma unroll
        for (int it = 0; it < 8; ++it) {
            int off = (tid + it * 128) * 16;
            async_load16(dstK + off, srcK + off);
        }
        unsigned dstV = lds_off32(&Vt[buf][0][0]);
#pragma unroll
        for (int it = 0; it < 8; ++it) {
            int task = tid + it * 128;           // 0..1023 : (c, 16B-part)
            int c = task >> 2, part = task & 3;
            const char* srcV = (const char*)(vv + (size_t)c * HW_ + j0) + part * 16;
            async_load16(dstV + task * 16, srcV);
        }
    };
    stage_kv(0, 0);
    wait_async0();
    __syncthreads();

    v8f O[16];
#pragma unroll
    for (int f = 0; f < 16; ++f) O[f] = zero8();
    float mi[8], li[8];
#pragma unroll
    for (int r = 0; r < 8; ++r) { mi[r] = -3.0e38f; li[r] = 0.f; }

    for (int jt = 0; jt < HW_ / 32; ++jt) {
        int cur = jt & 1;
        if (jt + 1 < HW_ / 32) stage_kv(cur ^ 1, (jt + 1) * 32);

        const _Float16* Kb = &Kt[cur][0][0];
        const _Float16* Vb = &Vt[cur][0][0];

        // S (16x32) = Q[wave] . K_tile^T, all from LDS
        v8f s0 = zero8(), s1 = zero8();
#pragma unroll
        for (int ks = 0; ks < C_; ks += 32) {
            v16h a  = frag_a(&Qt[0][0], C_, wid * 16, ks, lane);
            v16h b0 = frag_b(Kb, C_, 0,  ks, lane);
            v16h b1 = frag_b(Kb, C_, 16, ks, lane);
            s0 = wmma_f16(a, b0, s0);
            s1 = wmma_f16(a, b1, s1);
        }

        float mnew[8], alpha[8];
#pragma unroll
        for (int r = 0; r < 8; ++r) {
            s0[r] *= 0.0625f; s1[r] *= 0.0625f;          // 1/sqrt(C)
            float t = fmaxf(s0[r], s1[r]);
            t = fmaxf(t, __shfl_xor(t, 1));
            t = fmaxf(t, __shfl_xor(t, 2));
            t = fmaxf(t, __shfl_xor(t, 4));
            t = fmaxf(t, __shfl_xor(t, 8));              // row max within 16-lane half
            mnew[r]  = fmaxf(mi[r], t);
            alpha[r] = __expf(mi[r] - mnew[r]);
            mi[r] = mnew[r];
        }
#pragma unroll
        for (int r = 0; r < 8; ++r) {
            float p0 = __expf(s0[r] - mnew[r]);
            float p1 = __expf(s1[r] - mnew[r]);
            s0[r] = p0; s1[r] = p1;
            float t = p0 + p1;
            t += __shfl_xor(t, 1); t += __shfl_xor(t, 2);
            t += __shfl_xor(t, 4); t += __shfl_xor(t, 8);
            li[r] = li[r] * alpha[r] + t;
        }
#pragma unroll
        for (int f = 0; f < 16; ++f)
#pragma unroll
            for (int r = 0; r < 8; ++r) O[f][r] *= alpha[r];

        // P: C/D layout -> LDS row-major [16][32] -> reload in A layout
#pragma unroll
        for (int r = 0; r < 8; ++r) {
            Plds[wid][r + 8 * half][lr]      = (_Float16)s0[r];
            Plds[wid][r + 8 * half][16 + lr] = (_Float16)s1[r];
        }
        __syncthreads();
        v16h pa;
        {
            const _Float16* base = &Plds[wid][lr][half * 8];
            v8h lo = *(const v8h*)base;
            v8h hi = *(const v8h*)(base + 16);
#pragma unroll
            for (int i = 0; i < 8; ++i) { pa[i] = lo[i]; pa[i + 8] = hi[i]; }
        }

        // O(16x256) += P(16x32) . V(32x256), V from LDS [C][32]; batch 4 frags
#pragma unroll
        for (int f4 = 0; f4 < 16; f4 += 4) {
            v16h bv[4];
#pragma unroll
            for (int j = 0; j < 4; ++j) bv[j] = frag_b(Vb, 32, (f4 + j) * 16, 0, lane);
#pragma unroll
            for (int j = 0; j < 4; ++j) O[f4 + j] = wmma_f16(pa, bv[j], O[f4 + j]);
        }

        wait_async0();          // next-buffer fills done (issued before compute)
        __syncthreads();        // all waves done with `cur`, next buffer visible
    }

    float linv[8];
#pragma unroll
    for (int r = 0; r < 8; ++r) linv[r] = 1.f / li[r];
    _Float16* dst = outT + (size_t)b * HW_ * C_;
#pragma unroll
    for (int f = 0; f < 16; ++f) {
#pragma unroll
        for (int r = 0; r < 8; ++r) {
            int i = ib0 + wid * 16 + r + 8 * half;
            int c = f * 16 + lr;
            dst[(size_t)i * C_ + c] = (_Float16)(O[f][r] * linv[r]);
        }
    }
}

// ---------------- K5: proj GEMM + bias + residual (async-staged B tile) ----------------
__global__ __launch_bounds__(128) void proj_gemm_kernel(
        const _Float16* __restrict__ w16, const _Float16* __restrict__ attnT,
        const float* __restrict__ bias, const float* __restrict__ x,
        float* __restrict__ out) {
    int tid = threadIdx.x;
    int lane = tid & 31, wid = tid >> 5;
    int half = lane >> 4, lr = lane & 15;
    int n0 = blockIdx.x * 64;
    int m0 = blockIdx.y * 64 + wid * 16;
    int b  = blockIdx.z;
    const _Float16* ab = attnT + (size_t)b * HW_ * C_;

    __shared__ __align__(16) _Float16 Bt[64][C_];          // 32 KB
    {
        unsigned dst = lds_off32(&Bt[0][0]);
        const char* src = (const char*)(ab + (size_t)n0 * C_);
#pragma unroll
        for (int it = 0; it < 16; ++it) {
            int off = (tid + it * 128) * 16;
            async_load16(dst + off, src + off);
        }
    }
    wait_async0();
    __syncthreads();

    v8f acc[4];
#pragma unroll
    for (int f = 0; f < 4; ++f) acc[f] = zero8();

#pragma unroll
    for (int k0 = 0; k0 < C_; k0 += 32) {
        v16h a = frag_a(w16, C_, m0, k0, lane);
        v16h bb[4];
#pragma unroll
        for (int f = 0; f < 4; ++f) bb[f] = frag_b(&Bt[0][0], C_, f * 16, k0, lane);
#pragma unroll
        for (int f = 0; f < 4; ++f) acc[f] = wmma_f16(a, bb[f], acc[f]);
    }

    const float* xb = x   + (size_t)b * C_ * HW_;
    float*       ob = out + (size_t)b * C_ * HW_;
#pragma unroll
    for (int f = 0; f < 4; ++f) {
#pragma unroll
        for (int r = 0; r < 8; ++r) {
            int o = m0 + r + 8 * half;
            int n = n0 + f * 16 + lr;
            size_t idx = (size_t)o * HW_ + n;
            ob[idx] = xb[idx] + acc[f][r] + bias[o];
        }
    }
}

extern "C" void kernel_launch(void* const* d_in, const int* in_sizes, int n_in,
                              void* d_out, int out_size, void* d_ws, size_t ws_size,
                              hipStream_t stream) {
    const float* x      = (const float*)d_in[0];
    const float* gn_w   = (const float*)d_in[1];
    const float* gn_b   = (const float*)d_in[2];
    const float* qkv_w  = (const float*)d_in[3];
    const float* qkv_b  = (const float*)d_in[4];
    const float* proj_w = (const float*)d_in[5];
    const float* proj_b = (const float*)d_in[6];
    float* out = (float*)d_out;

    char* ws = (char*)d_ws;
    constexpr size_t SZ_T = (size_t)B_ * HW_ * C_ * 2;   // 16 MB per f16 tensor
    float*    mean = (float*)(ws);
    float*    rstd = (float*)(ws + 4096);
    _Float16* wq16 = (_Float16*)(ws + 8192);
    _Float16* wp16 = (_Float16*)(ws + 8192 + 393216);
    _Float16* xnT  = (_Float16*)(ws + 532480);
    _Float16* qT   = (_Float16*)(ws + 532480 + 1 * SZ_T);
    _Float16* kT   = (_Float16*)(ws + 532480 + 2 * SZ_T);
    _Float16* vM   = (_Float16*)(ws + 532480 + 3 * SZ_T);
    _Float16* aT   = (_Float16*)(ws + 532480 + 4 * SZ_T);

    cvt_weights_kernel<<<768, 256, 0, stream>>>(qkv_w, proj_w, wq16, wp16);
    gn_stats_kernel<<<B_ * G_, 256, 0, stream>>>(x, mean, rstd);
    gn_apply_kernel<<<dim3(HW_ / 64, C_ / 64, B_), 256, 0, stream>>>(x, gn_w, gn_b, mean, rstd, xnT);
    qkv_gemm_kernel<<<dim3(HW_ / 64, (3 * C_) / 64, B_), 128, 0, stream>>>(wq16, xnT, qkv_b, qT, kT, vM);
    attention_kernel<<<dim3(HW_ / 64, B_), 128, 0, stream>>>(qT, kT, vM, aT);
    proj_gemm_kernel<<<dim3(HW_ / 64, C_ / 64, B_), 128, 0, stream>>>(wp16, aT, proj_b, x, out);
}
